// EuclideanCodebook_2473901162732
// MI455X (gfx1250) — matrix operand order repaired and checked by reading, not compiled, not run on hardware
//
#include <hip/hip_runtime.h>
#include <hip/hip_bf16.h>

// ---------------------------------------------------------------------------
// VQ nearest-codebook search on gfx1250 (MI455X), bf16-split WMMA GEMM.
//   argmin_k (b2[k] - 2 * x.e_k)        (||x||^2 is row-constant -> dropped)
//   dot = hi*hi + hi*lo + lo*hi, three independent v_wmma accumulator chains
//   B tiles staged LDS-side by the Tensor Data Mover (double-buffered,
//   TENSORcnt-tracked), with D# LDS padding for conflict-free 544B rows.
// ---------------------------------------------------------------------------

typedef __attribute__((ext_vector_type(16))) __bf16   v16bf;
typedef __attribute__((ext_vector_type(8)))  __bf16   v8bf;
typedef __attribute__((ext_vector_type(8)))  float    v8f;
typedef __attribute__((ext_vector_type(4)))  unsigned v4u;
typedef __attribute__((ext_vector_type(8)))  int      v8i;
typedef __attribute__((ext_vector_type(4)))  int      v4i;

#define DIM 256

__device__ __forceinline__ __bf16 f2bf(float f) {
    unsigned u = __builtin_bit_cast(unsigned, f);
    u += 0x7FFFu + ((u >> 16) & 1u);          // round-to-nearest-even
    unsigned short s = (unsigned short)(u >> 16);
    return __builtin_bit_cast(__bf16, s);
}
__device__ __forceinline__ float bf2f(__bf16 b) {
    unsigned u = ((unsigned)__builtin_bit_cast(unsigned short, b)) << 16;
    return __builtin_bit_cast(float, u);
}

__device__ __forceinline__ v8f wmma_bf16(v16bf a, v16bf b, v8f c) {
    return __builtin_amdgcn_wmma_f32_16x16x32_bf16(false, a, false, b,
                                                   (short)0, c, false, false);
}

// ---------------------------------------------------------------------------
// TDM: load a 16-row x 512-byte tile (bf16 16x256) into LDS with 32B row pad.
// D# built per CDNA5 ISA (2-D tensor, groups 2/3 NULL).
// This toolchain exposes the 6-arg builtin:
//   (uint32x4 g0, int32x8 g1, int32x4 g2, int32x4 g3, int32x8 g4, i32 cpol)
// ---------------------------------------------------------------------------
__device__ __forceinline__ void tdm_load_tile(unsigned ldsOff,
                                              unsigned long long gaddr,
                                              int K) {
    v4u g0;
    g0[0] = 1u;                                    // count=1, user descriptor
    g0[1] = ldsOff;                                // lds_addr (bytes)
    g0[2] = (unsigned)(gaddr & 0xFFFFFFFFull);     // global_addr[31:0]
    g0[3] = (unsigned)((gaddr >> 32) & 0x01FFFFFFull) | (2u << 30); // type=2

    v8i g1;
    //   data_size=1 (2B) | pad_enable | pad_interval=6 (128 DW = 512B)
    // | pad_amount=7 (8 DW = 32B)
    g1[0] = (1 << 16) | (1 << 20) | (6 << 22) | (7 << 25);
    g1[1] = (int)(256u << 16);                     // tensor_dim0 = 256 elems
    g1[2] = (int)(((unsigned)K & 0xFFFFu) << 16);  // tensor_dim1 = K
    g1[3] = (int)(256u << 16);                     // tile_dim0 = 256 elems
    g1[4] = 16;                                    // tile_dim1 = 16 rows
    g1[5] = 256;                                   // tensor_dim0_stride = 256
    g1[6] = 0;
    g1[7] = 0;

    v4i z4 = {0, 0, 0, 0};                         // 2-D: groups 2/3 unused
    v8i z8 = {0, 0, 0, 0, 0, 0, 0, 0};             // extra group (unused)
    __builtin_amdgcn_tensor_load_to_lds(g0, g1, z4, z4, z8, 0);
}

__device__ __forceinline__ unsigned lds_offset(const void* p) {
    // generic LDS pointer: ADDR[31:0] is the LDS byte address
    return (unsigned)(unsigned long long)p;
}

// ---------------------------------------------------------------------------
// Kernel 1: embedding -> bf16 hi/lo split + ||e_k||^2.  One wave32 per row.
// ---------------------------------------------------------------------------
__global__ __launch_bounds__(256)
void vq_prep_kernel(const float* __restrict__ emb,
                    __bf16* __restrict__ ebHi, __bf16* __restrict__ ebLo,
                    float* __restrict__ b2, int K) {
    const int gw   = (blockIdx.x * blockDim.x + threadIdx.x) >> 5; // row k
    const int lane = threadIdx.x & 31;
    if (gw >= K) return;

    const float4* src = (const float4*)(emb + (size_t)gw * DIM + lane * 8);
    float4 p0 = src[0], p1 = src[1];
    float v[8] = {p0.x, p0.y, p0.z, p0.w, p1.x, p1.y, p1.z, p1.w};

    v8bf h, l;
    float ss = 0.f;
#pragma unroll
    for (int i = 0; i < 8; ++i) {
        __bf16 hb = f2bf(v[i]);
        h[i] = hb;
        l[i] = f2bf(v[i] - bf2f(hb));
        ss  += v[i] * v[i];
    }
    *(v8bf*)(ebHi + (size_t)gw * DIM + lane * 8) = h;
    *(v8bf*)(ebLo + (size_t)gw * DIM + lane * 8) = l;

#pragma unroll
    for (int m = 16; m >= 1; m >>= 1) ss += __shfl_xor(ss, m, 32);
    if (lane == 0) b2[gw] = ss;
}

// ---------------------------------------------------------------------------
// Kernel 2: main search. Block = 8 waves * 16 rows = 128 rows.
// B tile (16 codes, hi+lo) TDM-staged into double-buffered LDS.
// ---------------------------------------------------------------------------
#define LDS_ROW 272   // 256 bf16 + 16 pad -> 544B row stride, conflict-free

__global__ __launch_bounds__(256)
void vq_search_kernel(const float* __restrict__ x,
                      const float* __restrict__ emb,
                      const __bf16* __restrict__ ebHi,
                      const __bf16* __restrict__ ebLo,
                      const float* __restrict__ b2,
                      float* __restrict__ quant,
                      float* __restrict__ xf,
                      int* __restrict__ codes,
                      int K) {
    const int tid  = threadIdx.x;
    const int lane = tid & 31;
    const int wave = tid >> 5;
    const int half = lane >> 4;         // 0: lanes 0-15, 1: lanes 16-31
    const int r    = lane & 15;         // row-in-tile (A) / code-in-tile (B)
    const int rowBase = blockIdx.x * 128 + wave * 16;
    const int row     = rowBase + r;

    __shared__ __align__(32) __bf16 sHi[2][16 * LDS_ROW];
    __shared__ __align__(32) __bf16 sLo[2][16 * LDS_ROW];
    __shared__ float sB2[2048];

    for (int i = tid; i < K; i += 256) sB2[i] = b2[i];

    // ---- Build A fragments (16 rows x 256 dims, hi+lo) and stream xf copy.
    // A layout (matches B): lanes 0-15 hold K c*32+0..15,
    // lanes 16-31 hold K c*32+16..31 for chunk c.
    v16bf aHi[8], aLo[8];
#pragma unroll
    for (int c = 0; c < 8; ++c) {
        const int d0 = c * 32 + half * 16;
        const int d1 = c * 32 + half * 16 + 8;
        const float4* p0 = (const float4*)(x + (size_t)row * DIM + d0);
        const float4* p1 = (const float4*)(x + (size_t)row * DIM + d1);
        float4 q0 = p0[0], q1 = p0[1], q2 = p1[0], q3 = p1[1];

        float4* o0 = (float4*)(xf + (size_t)row * DIM + d0);
        float4* o1 = (float4*)(xf + (size_t)row * DIM + d1);
        o0[0] = q0; o0[1] = q1; o1[0] = q2; o1[1] = q3;

        float v[16] = {q0.x, q0.y, q0.z, q0.w, q1.x, q1.y, q1.z, q1.w,
                       q2.x, q2.y, q2.z, q2.w, q3.x, q3.y, q3.z, q3.w};
#pragma unroll
        for (int i = 0; i < 16; ++i) {
            __bf16 hb = f2bf(v[i]);
            aHi[c][i] = hb;
            aLo[c][i] = f2bf(v[i] - bf2f(hb));
        }
    }

    float best[8];
    int   bidx[8];
#pragma unroll
    for (int j = 0; j < 8; ++j) { best[j] = 3.402823466e38f; bidx[j] = 0; }

    const int nTiles = K >> 4;

    // ---- TDM prologue: wave 0 stages tile 0 into buffer 0
    if (wave == 0) {
        tdm_load_tile(lds_offset(&sHi[0][0]), (unsigned long long)(ebHi), K);
        tdm_load_tile(lds_offset(&sLo[0][0]), (unsigned long long)(ebLo), K);
    }

    for (int ct = 0; ct < nTiles; ++ct) {
        const int buf = ct & 1;
        __syncthreads();   // all waves done reading buf^1 (iteration ct-1)
        if (wave == 0) {
            if (ct + 1 < nTiles) {   // prefetch next tile into the free buffer
                const size_t gof = (size_t)(ct + 1) * 16 * DIM * sizeof(__bf16);
                tdm_load_tile(lds_offset(&sHi[buf ^ 1][0]),
                              (unsigned long long)ebHi + gof, K);
                tdm_load_tile(lds_offset(&sLo[buf ^ 1][0]),
                              (unsigned long long)ebLo + gof, K);
                __builtin_amdgcn_s_wait_tensorcnt(2);  // tile ct landed
            } else {
                __builtin_amdgcn_s_wait_tensorcnt(0);
            }
        }
        __syncthreads();   // tile ct visible to all waves

        // B layout used here (same k order as A): lane holds column n=r, K = half*16 + 0..15
        const __bf16* bHiRow = &sHi[buf][r * LDS_ROW + half * 16];
        const __bf16* bLoRow = &sLo[buf][r * LDS_ROW + half * 16];
        v8f accHH = {}, accHL = {}, accLH = {};   // 3 independent chains
#pragma unroll
        for (int c = 0; c < 8; ++c) {
            const v16bf bh = *(const v16bf*)(bHiRow + c * 32);
            const v16bf bl = *(const v16bf*)(bLoRow + c * 32);
            accHH = wmma_bf16(aHi[c], bh, accHH);
            accHL = wmma_bf16(aHi[c], bl, accHL);
            accLH = wmma_bf16(aLo[c], bh, accLH);
        }

        const int n    = ct * 16 + r;    // candidate code for this lane
        const float bn = sB2[n];
#pragma unroll
        for (int j = 0; j < 8; ++j) {    // C/D layout: VGPR j = row j + 8*half
            float dot = accHH[j] + accHL[j] + accLH[j];
            float s   = fmaf(-2.0f, dot, bn);
            if (s < best[j]) { best[j] = s; bidx[j] = n; }
        }
    }

    // ---- argmin across the 16 lanes of each half (wave32 shuffles)
#pragma unroll
    for (int m = 8; m >= 1; m >>= 1) {
#pragma unroll
        for (int j = 0; j < 8; ++j) {
            float os = __shfl_xor(best[j], m, 32);
            int   oi = __shfl_xor(bidx[j], m, 32);
            if (os < best[j] || (os == best[j] && oi < bidx[j])) {
                best[j] = os; bidx[j] = oi;
            }
        }
    }

    if (r == 0) {
#pragma unroll
        for (int j = 0; j < 8; ++j) codes[rowBase + half * 8 + j] = bidx[j];
    }

    // ---- gather quantized = embedding[code] (f32), 32 lanes per row
#pragma unroll
    for (int m = 0; m < 16; ++m) {
        const int j   = m & 7;
        const int src = (m < 8) ? 0 : 16;
        const int cm  = __shfl(bidx[j], src, 32);
        const float4* ep = (const float4*)(emb + (size_t)cm * DIM + lane * 8);
        float4 e0 = ep[0], e1 = ep[1];
        float4* qp = (float4*)(quant + (size_t)(rowBase + m) * DIM + lane * 8);
        qp[0] = e0; qp[1] = e1;
    }
}

// ---------------------------------------------------------------------------
extern "C" void kernel_launch(void* const* d_in, const int* in_sizes, int n_in,
                              void* d_out, int out_size, void* d_ws, size_t ws_size,
                              hipStream_t stream) {
    const float* x   = (const float*)d_in[0];
    const float* emb = (const float*)d_in[1];
    const int N = in_sizes[0] / DIM;     // 32768
    const int K = in_sizes[1] / DIM;     // 2048

    // workspace: [ ebHi: K*256 bf16 | ebLo: K*256 bf16 | b2: K f32 ]
    char* ws = (char*)d_ws;
    __bf16* ebHi = (__bf16*)ws;
    __bf16* ebLo = (__bf16*)(ws + (size_t)K * DIM * 2);
    float*  b2   = (float*)(ws + (size_t)K * DIM * 4);

    // outputs: quantized [N,256] f32 | xf [N,256] f32 | codes [N] int32
    float* quant = (float*)d_out;
    float* xf    = quant + (size_t)N * DIM;
    int*   codes = (int*)(xf + (size_t)N * DIM);

    vq_prep_kernel<<<(K + 7) / 8, 256, 0, stream>>>(emb, ebHi, ebLo, b2, K);
    vq_search_kernel<<<N / 128, 256, 0, stream>>>(x, emb, ebHi, ebLo, b2,
                                                  quant, xf, codes, K);
}
